// Net_60138132078719
// MI455X (gfx1250) — matrix operand order repaired and checked by reading, not compiled, not run on hardware
//
#include <hip/hip_runtime.h>

namespace {
constexpr int PD = 28;            // hidden dim
constexpr int NN = 25000;         // nodes
constexpr int NE = 250000;        // edges
constexpr int NB = 512;           // graphs
constexpr float BN_EPS = 1e-5f;
constexpr int ETILES = NE / 16;   // 15625 exact
constexpr int MT_ELEMS = 32 * 896;            // bf16 elements in folded weight matrix
constexpr int MT_BYTES = MT_ELEMS * 2;        // 57344 bytes (dynamic LDS per block)
}

typedef __attribute__((ext_vector_type(16))) __bf16 v16bf;
typedef __attribute__((ext_vector_type(8)))  float  v8f;

// ---------------- helpers ----------------
__device__ __forceinline__ float lrelu(float x) { return x > 0.f ? x : 0.01f * x; }
__device__ __forceinline__ float sigm(float x)  { return 1.f / (1.f + __expf(-x)); }

__device__ __forceinline__ unsigned short fbits_bf16(float f) {
  union { __bf16 b; unsigned short u; } c; c.b = (__bf16)f; return c.u;
}
__device__ __forceinline__ float bflo(unsigned u) {
  union { unsigned u; float f; } c; c.u = (u & 0xffffu) << 16; return c.f;
}
__device__ __forceinline__ float bfhi(unsigned u) {
  union { unsigned u; float f; } c; c.u = u & 0xffff0000u; return c.f;
}
__device__ __forceinline__ unsigned encf(float f) {   // monotone float->uint
  unsigned u = __float_as_uint(f);
  return (u & 0x80000000u) ? ~u : (u | 0x80000000u);
}
__device__ __forceinline__ float decf(unsigned u) {
  unsigned v = (u & 0x80000000u) ? (u & 0x7fffffffu) : ~u;
  return __uint_as_float(v);
}
__device__ __forceinline__ void atomAddF(float* p, float v) {
  __hip_atomic_fetch_add(p, v, __ATOMIC_RELAXED, __HIP_MEMORY_SCOPE_AGENT);
}

// ---------------- small prep kernels ----------------

// Mt[n][k]  n in 0..31 (output o, padded), k = i*32+j (i in 0..27, j in 0..31)
//  j<28 : s_{i,n} * W2[i*28+n, j]     (BN scale folded)
//  j==28: b2 - m2*s  (BN shift; pairs with hp[:,28]==1)
//  else : 0
__global__ void k_prep_m(const float* __restrict__ w2, const float* __restrict__ g2,
                         const float* __restrict__ b2, const float* __restrict__ m2,
                         const float* __restrict__ v2, unsigned short* __restrict__ Mt) {
  int t = blockIdx.x * blockDim.x + threadIdx.x;
  if (t >= MT_ELEMS) return;
  int n = t / 896, k = t % 896;
  int i = k >> 5, j = k & 31;
  float val = 0.f;
  if (n < PD) {
    int ko = i * PD + n;
    float s = g2[ko] * rsqrtf(v2[ko] + BN_EPS);
    if (j < PD)       val = s * w2[(size_t)ko * PD + j];
    else if (j == PD) val = b2[ko] - m2[ko] * s;
  }
  Mt[(size_t)n * 896 + k] = fbits_bf16(val);
}

// out0 = lrelu(bn0(x @ lin0_w.T)); writes fp32 (stride 28) and bf16 (stride 32, padded)
__global__ void k_embed(const float* __restrict__ x, const float* __restrict__ w,
                        const float* __restrict__ g, const float* __restrict__ b,
                        const float* __restrict__ mm, const float* __restrict__ vv,
                        float* __restrict__ outf, unsigned short* __restrict__ outb) {
  int t = blockIdx.x * blockDim.x + threadIdx.x;
  if (t >= NN * 32) return;
  int nd = t >> 5, o = t & 31;
  if (o >= PD) { outb[(size_t)nd * 32 + o] = 0; return; }
  const float* xr = x + (size_t)nd * 16;
  const float* wr = w + o * 16;
  float acc = 0.f;
#pragma unroll
  for (int i = 0; i < 16; ++i) acc += xr[i] * wr[i];
  acc = lrelu(g[o] * (acc - mm[o]) * rsqrtf(vv[o] + BN_EPS) + b[o]);
  outf[(size_t)nd * PD + o] = acc;
  outb[(size_t)nd * 32 + o] = fbits_bf16(acc);
}

// hp[e][j]: j<28 -> lrelu(bn1(edge_attr @ W1.T)); j==28 -> 1; j>28 -> 0   (bf16)
__global__ void k_edge1(const float* __restrict__ ea, const float* __restrict__ w1,
                        const float* __restrict__ g, const float* __restrict__ b,
                        const float* __restrict__ mm, const float* __restrict__ vv,
                        unsigned short* __restrict__ hp) {
  int t = blockIdx.x * blockDim.x + threadIdx.x;
  if (t >= NE * 32) return;
  int e = t >> 5, j = t & 31;
  float val;
  if (j < PD) {
    const float* er = ea + (size_t)e * 8;
    const float* wr = w1 + j * 8;
    float acc = 0.f;
#pragma unroll
    for (int i = 0; i < 8; ++i) acc += er[i] * wr[i];
    val = lrelu(g[j] * (acc - mm[j]) * rsqrtf(vv[j] + BN_EPS) + b[j]);
  } else {
    val = (j == PD) ? 1.f : 0.f;
  }
  hp[(size_t)e * 32 + j] = fbits_bf16(val);
}

__global__ void k_deg(const int* __restrict__ dst, float* __restrict__ deg) {
  int e = blockIdx.x * blockDim.x + threadIdx.x;
  if (e >= NE) return;
  atomAddF(&deg[dst[e]], 1.f);
}

// ---------------- WMMA fused message kernel ----------------
// msg[e,o] = sum_{i,j} out_bf[src[e],i] * hp[e,j] * Mt[o][i*32+j]
// One wave per 16-edge tile; 28 K-chunks of 32; 2 N-tiles -> 56 v_wmma per wave.
// Mt (57 KB of folded weights) is staged in LDS once per workgroup. B fragments
// are prefetched TWO chunks ahead (triple buffer) so s_wait_dscnt retires early,
// and the A fragment alternates between two register sets so the next chunk's
// A-build VALU does not WAR-hazard the in-flight WMMA (no v_nop insertion).
// Results atomically scattered into agg[dst[e]*28 + o].
__global__ __launch_bounds__(256) void k_msg(const unsigned short* __restrict__ hp,
                                             const unsigned short* __restrict__ outbf,
                                             const unsigned short* __restrict__ Mt,
                                             const int* __restrict__ src,
                                             const int* __restrict__ dst,
                                             float* __restrict__ agg) {
  extern __shared__ char smem[];
  unsigned short* sMt = (unsigned short*)smem;

  // cooperative staging: 7168 uint4 total, 28 per thread
  {
    const uint4* gsrc = (const uint4*)Mt;
    uint4* ldst = (uint4*)sMt;
#pragma unroll
    for (int r = 0; r < 28; ++r) {
      int idx = r * 256 + (int)threadIdx.x;
      ldst[idx] = gsrc[idx];
    }
  }
  __syncthreads();

  const int wave = (blockIdx.x * blockDim.x + threadIdx.x) >> 5;
  const int lane = threadIdx.x & 31;
  if (wave >= ETILES) return;
  const int e0   = wave * 16;
  const int m    = lane & 15;
  const int half = lane >> 4;
  const int em   = e0 + m;

  // speculative prefetch of next tile's hp rows
  __builtin_prefetch(hp + (size_t)(em + 16) * 32, 0, 1);

  // source row: 28 bf16 (padded to 32) -> 16 dwords
  const int s = src[em];
  const uint4* ap = (const uint4*)(outbf + (size_t)s * 32);
  uint4 a0 = ap[0], a1 = ap[1], a2 = ap[2], a3 = ap[3];
  unsigned au[16] = {a0.x, a0.y, a0.z, a0.w, a1.x, a1.y, a1.z, a1.w,
                     a2.x, a2.y, a2.z, a2.w, a3.x, a3.y, a3.z, a3.w};

  // hp fragment in A layout: e[t] <-> j = t + 16*half
  const uint4 h0 = *(const uint4*)(hp + (size_t)em * 32 + half * 16);
  const uint4 h1 = *(const uint4*)(hp + (size_t)em * 32 + half * 16 + 8);
  float hf[16] = {bflo(h0.x), bfhi(h0.x), bflo(h0.y), bfhi(h0.y),
                  bflo(h0.z), bfhi(h0.z), bflo(h0.w), bfhi(h0.w),
                  bflo(h1.x), bfhi(h1.x), bflo(h1.y), bfhi(h1.y),
                  bflo(h1.z), bfhi(h1.z), bflo(h1.w), bfhi(h1.w)};

  v8f acc0 = {0.f, 0.f, 0.f, 0.f, 0.f, 0.f, 0.f, 0.f};
  v8f acc1 = {0.f, 0.f, 0.f, 0.f, 0.f, 0.f, 0.f, 0.f};

  const unsigned short* MtA = sMt + (size_t)(lane & 15) * 896 + half * 16;
  const unsigned short* MtB = sMt + (size_t)((lane & 15) + 16) * 896 + half * 16;

  // triple-buffered B fragments out of LDS (prefetch distance 2)
  v16bf b0c = *(const v16bf*)(MtA);
  v16bf b1c = *(const v16bf*)(MtB);
  v16bf b0n = *(const v16bf*)(MtA + 32);
  v16bf b1n = *(const v16bf*)(MtB + 32);

  // two alternating A-fragment register sets (avoids WMMA WAR hazard NOPs)
  union AF { v16bf v; __bf16 e[16]; };
  AF Abuf[2];

#pragma unroll
  for (int i = 0; i < PD; ++i) {
    // issue loads for chunk i+2 first: ~1 full iteration of latency slack
    v16bf t0, t1;
    if (i + 2 < PD) {
      t0 = *(const v16bf*)(MtA + (i + 2) * 32);
      t1 = *(const v16bf*)(MtB + (i + 2) * 32);
    }
    unsigned pa = au[i >> 1];
    float ai = (i & 1) ? bfhi(pa) : bflo(pa);
    AF& A = Abuf[i & 1];
#pragma unroll
    for (int t = 0; t < 16; ++t) A.e[t] = (__bf16)(ai * hf[t]);
    acc0 = __builtin_amdgcn_wmma_f32_16x16x32_bf16(false, A.v, false, b0c,
                                                   (short)0, acc0, false, false);
    acc1 = __builtin_amdgcn_wmma_f32_16x16x32_bf16(false, A.v, false, b1c,
                                                   (short)0, acc1, false, false);
    b0c = b0n; b1c = b1n;
    b0n = t0;  b1n = t1;
  }

  // D layout: element g <-> (M = g + 8*half, N = lane&15)
  const int n = lane & 15;
#pragma unroll
  for (int g = 0; g < 8; ++g) {
    int row = half * 8 + g;
    int d = dst[e0 + row];
    float* base = agg + (size_t)d * PD;
    atomAddF(base + n, acc0[g]);
    if (n + 16 < PD) atomAddF(base + n + 16, acc1[g]);
  }
}

// ---------------- NNConv epilogue + GRU (wave per node) ----------------
__global__ __launch_bounds__(256) void k_update(float* __restrict__ outf,
                                                unsigned short* __restrict__ outb,
                                                const float* __restrict__ agg,
                                                const float* __restrict__ deg,
                                                const float* __restrict__ root,
                                                const float* __restrict__ convb,
                                                const float* __restrict__ wih,
                                                const float* __restrict__ whh,
                                                const float* __restrict__ bih,
                                                const float* __restrict__ bhh) {
  int wave = (blockIdx.x * blockDim.x + threadIdx.x) >> 5;
  int lane = threadIdx.x & 31;
  if (wave >= NN) return;
  int nd = wave;
  float h = (lane < PD) ? outf[(size_t)nd * PD + lane] : 0.f;
  float mv = 0.f;
  if (lane < PD)
    mv = agg[(size_t)nd * PD + lane] / fmaxf(deg[nd], 1.f) + convb[lane];
  for (int i = 0; i < PD; ++i) {
    float hi = __shfl(h, i, 32);
    if (lane < PD) mv += hi * root[i * PD + lane];
  }
  mv = lrelu(mv);
  float gr = 0.f, gz = 0.f, gn = 0.f, hr = 0.f, hz = 0.f, hn = 0.f;
  if (lane < PD) {
    gr = bih[lane]; gz = bih[PD + lane]; gn = bih[2 * PD + lane];
    hr = bhh[lane]; hz = bhh[PD + lane]; hn = bhh[2 * PD + lane];
  }
  for (int i = 0; i < PD; ++i) {
    float mi = __shfl(mv, i, 32);
    float hi = __shfl(h, i, 32);
    if (lane < PD) {
      gr += mi * wih[lane * PD + i];
      gz += mi * wih[(PD + lane) * PD + i];
      gn += mi * wih[(2 * PD + lane) * PD + i];
      hr += hi * whh[lane * PD + i];
      hz += hi * whh[(PD + lane) * PD + i];
      hn += hi * whh[(2 * PD + lane) * PD + i];
    }
  }
  if (lane < PD) {
    float r = sigm(gr + hr), z = sigm(gz + hz);
    float nn = tanhf(gn + r * hn);
    float hnew = (1.f - z) * nn + z * h;
    outf[(size_t)nd * PD + lane] = hnew;
    outb[(size_t)nd * 32 + lane] = fbits_bf16(hnew);
  }
}

// ---------------- Set2Set ----------------
__global__ __launch_bounds__(256) void k_lstm(const float* __restrict__ qs,
                                              float* __restrict__ hh, float* __restrict__ cc,
                                              const float* __restrict__ wih,
                                              const float* __restrict__ whh,
                                              const float* __restrict__ bih,
                                              const float* __restrict__ bhh) {
  int wave = (blockIdx.x * blockDim.x + threadIdx.x) >> 5;
  int lane = threadIdx.x & 31;
  if (wave >= NB) return;
  int gph = wave;
  float hprev = (lane < PD) ? hh[gph * PD + lane] : 0.f;
  float gi = 0.f, gf = 0.f, gg = 0.f, go = 0.f;
  if (lane < PD) {
    int r0 = lane, r1 = PD + lane, r2 = 2 * PD + lane, r3 = 3 * PD + lane;
    gi = bih[r0] + bhh[r0]; gf = bih[r1] + bhh[r1];
    gg = bih[r2] + bhh[r2]; go = bih[r3] + bhh[r3];
    const float* q = qs + gph * (2 * PD);
    for (int j = 0; j < 2 * PD; ++j) {
      float qj = q[j];
      gi += qj * wih[r0 * 2 * PD + j]; gf += qj * wih[r1 * 2 * PD + j];
      gg += qj * wih[r2 * 2 * PD + j]; go += qj * wih[r3 * 2 * PD + j];
    }
  }
  for (int j = 0; j < PD; ++j) {
    float hj = __shfl(hprev, j, 32);
    if (lane < PD) {
      gi += hj * whh[lane * PD + j];
      gf += hj * whh[(PD + lane) * PD + j];
      gg += hj * whh[(2 * PD + lane) * PD + j];
      go += hj * whh[(3 * PD + lane) * PD + j];
    }
  }
  if (lane < PD) {
    float c = sigm(gf) * cc[gph * PD + lane] + sigm(gi) * tanhf(gg);
    cc[gph * PD + lane] = c;
    hh[gph * PD + lane] = sigm(go) * tanhf(c);
  }
}

__global__ __launch_bounds__(256) void k_att1(const float* __restrict__ outf,
                                              const float* __restrict__ hh,
                                              const int* __restrict__ batch,
                                              float* __restrict__ ebuf,
                                              unsigned* __restrict__ emax) {
  int wave = (blockIdx.x * blockDim.x + threadIdx.x) >> 5;
  int lane = threadIdx.x & 31;
  if (wave >= NN) return;
  int nd = wave, g = batch[nd];
  float p = (lane < PD) ? outf[(size_t)nd * PD + lane] * hh[g * PD + lane] : 0.f;
  for (int off = 16; off; off >>= 1) p += __shfl_xor(p, off, 32);
  if (lane == 0) {
    ebuf[nd] = p;
    atomicMax(emax + g, encf(p));
  }
}

__global__ void k_att2(const float* __restrict__ ebuf, const unsigned* __restrict__ emax,
                       const int* __restrict__ batch, float* __restrict__ abuf,
                       float* __restrict__ esum) {
  int nd = blockIdx.x * blockDim.x + threadIdx.x;
  if (nd >= NN) return;
  int g = batch[nd];
  float a = __expf(ebuf[nd] - decf(emax[g]));
  abuf[nd] = a;
  atomAddF(&esum[g], a);
}

__global__ __launch_bounds__(256) void k_att3(const float* __restrict__ outf,
                                              const float* __restrict__ abuf,
                                              const float* __restrict__ esum,
                                              const int* __restrict__ batch,
                                              float* __restrict__ rvec) {
  int wave = (blockIdx.x * blockDim.x + threadIdx.x) >> 5;
  int lane = threadIdx.x & 31;
  if (wave >= NN) return;
  int nd = wave, g = batch[nd];
  if (lane < PD) {
    float w = abuf[nd] / esum[g];
    atomAddF(&rvec[g * PD + lane], w * outf[(size_t)nd * PD + lane]);
  }
}

__global__ void k_qstar(const float* __restrict__ hh, const float* __restrict__ rvec,
                        float* __restrict__ qs) {
  int t = blockIdx.x * blockDim.x + threadIdx.x;
  if (t >= NB * 2 * PD) return;
  int g = t / (2 * PD), j = t % (2 * PD);
  qs[t] = (j < PD) ? hh[g * PD + j] : rvec[g * PD + (j - PD)];
}

// ---------------- final MLP head (wave per graph) ----------------
__global__ __launch_bounds__(256) void k_head(const float* __restrict__ qs,
                                              const float* __restrict__ l1w, const float* __restrict__ l1b,
                                              const float* __restrict__ l2w, const float* __restrict__ l2b,
                                              const float* __restrict__ lfw, const float* __restrict__ lfb,
                                              float* __restrict__ dout) {
  int wave = (blockIdx.x * blockDim.x + threadIdx.x) >> 5;
  int lane = threadIdx.x & 31;
  if (wave >= NB) return;
  int g = wave;
  float y1 = 0.f;
  if (lane < PD) {
    y1 = l1b[lane];
    const float* q = qs + g * (2 * PD);
    for (int j = 0; j < 2 * PD; ++j) y1 += q[j] * l1w[lane * 2 * PD + j];
    y1 = lrelu(y1);
  }
  float y2 = (lane < PD / 2) ? l2b[lane] : 0.f;
  for (int j = 0; j < PD; ++j) {
    float yj = __shfl(y1, j, 32);
    if (lane < PD / 2) y2 += yj * l2w[lane * PD + j];
  }
  y2 = (lane < PD / 2) ? lrelu(y2) : 0.f;
  float contrib = (lane < PD / 2) ? y2 * lfw[lane] : 0.f;
  for (int off = 16; off; off >>= 1) contrib += __shfl_xor(contrib, off, 32);
  if (lane == 0) dout[g] = contrib + lfb[0];
}

// ---------------- host launcher ----------------
extern "C" void kernel_launch(void* const* d_in, const int* in_sizes, int n_in,
                              void* d_out, int out_size, void* d_ws, size_t ws_size,
                              hipStream_t stream) {
  (void)in_sizes; (void)n_in; (void)out_size; (void)ws_size;
  const float* x    = (const float*)d_in[0];
  const int*   ei   = (const int*)d_in[1];
  const float* ea   = (const float*)d_in[2];
  const int*   batch= (const int*)d_in[3];
  const float* lin0_w = (const float*)d_in[4];
  const float* bn0g = (const float*)d_in[5],  *bn0b = (const float*)d_in[6];
  const float* bn0m = (const float*)d_in[7],  *bn0v = (const float*)d_in[8];
  const float* w1   = (const float*)d_in[9];
  const float* bn1g = (const float*)d_in[10], *bn1b = (const float*)d_in[11];
  const float* bn1m = (const float*)d_in[12], *bn1v = (const float*)d_in[13];
  const float* w2   = (const float*)d_in[14];
  const float* bn2g = (const float*)d_in[15], *bn2b = (const float*)d_in[16];
  const float* bn2m = (const float*)d_in[17], *bn2v = (const float*)d_in[18];
  const float* root = (const float*)d_in[19];
  const float* convb= (const float*)d_in[20];
  const float* gwih = (const float*)d_in[21], *gwhh = (const float*)d_in[22];
  const float* gbih = (const float*)d_in[23], *gbhh = (const float*)d_in[24];
  const float* lwih = (const float*)d_in[25], *lwhh = (const float*)d_in[26];
  const float* lbih = (const float*)d_in[27], *lbhh = (const float*)d_in[28];
  const float* l1w  = (const float*)d_in[29], *l1b  = (const float*)d_in[30];
  const float* l2w  = (const float*)d_in[31], *l2b  = (const float*)d_in[32];
  const float* lfw  = (const float*)d_in[33], *lfb  = (const float*)d_in[34];

  const int* src = ei;
  const int* dst = ei + NE;

  // workspace carve-up (256B aligned slices)
  char* base = (char*)d_ws;
  size_t off = 0;
  auto carve = [&](size_t bytes) -> void* {
    void* p = base + off;
    off = (off + bytes + 255) & ~(size_t)255;
    return p;
  };
  float*          outf = (float*)carve((size_t)NN * PD * 4);
  float*          agg  = (float*)carve((size_t)NN * PD * 4);
  float*          deg  = (float*)carve((size_t)NN * 4);
  unsigned short* outb = (unsigned short*)carve((size_t)NN * 32 * 2);
  unsigned short* hp   = (unsigned short*)carve((size_t)NE * 32 * 2);
  unsigned short* Mt   = (unsigned short*)carve((size_t)MT_BYTES);
  float*          ebuf = (float*)carve((size_t)NN * 4);
  float*          abuf = (float*)carve((size_t)NN * 4);
  // contiguous zero-block: emax(u32) | esum | rvec
  char*  attblk = (char*)carve((size_t)(NB + NB + NB * PD) * 4);
  unsigned* emax = (unsigned*)attblk;
  float*    esum = (float*)(attblk + NB * 4);
  float*    rvec = (float*)(attblk + 2 * NB * 4);
  float*    qs   = (float*)carve((size_t)NB * 2 * PD * 4);
  float*    hh   = (float*)carve((size_t)NB * PD * 4);
  float*    cc   = (float*)carve((size_t)NB * PD * 4);

  // one-time zero inits
  hipMemsetAsync(deg, 0, (size_t)NN * 4, stream);
  hipMemsetAsync(qs, 0, (size_t)NB * 2 * PD * 4, stream);
  hipMemsetAsync(hh, 0, (size_t)NB * PD * 4, stream);
  hipMemsetAsync(cc, 0, (size_t)NB * PD * 4, stream);

  k_prep_m<<<(MT_ELEMS + 255) / 256, 256, 0, stream>>>(w2, bn2g, bn2b, bn2m, bn2v, Mt);
  k_embed<<<(NN * 32 + 255) / 256, 256, 0, stream>>>(x, lin0_w, bn0g, bn0b, bn0m, bn0v, outf, outb);
  k_edge1<<<(NE * 32 + 255) / 256, 256, 0, stream>>>(ea, w1, bn1g, bn1b, bn1m, bn1v, hp);
  k_deg<<<(NE + 255) / 256, 256, 0, stream>>>(dst, deg);

  for (int t = 0; t < 3; ++t) {
    hipMemsetAsync(agg, 0, (size_t)NN * PD * 4, stream);
    k_msg<<<(ETILES * 32 + 255) / 256, 256, MT_BYTES, stream>>>(hp, outb, Mt, src, dst, agg);
    k_update<<<(NN * 32 + 255) / 256, 256, 0, stream>>>(outf, outb, agg, deg, root, convb,
                                                        gwih, gwhh, gbih, gbhh);
  }

  for (int s = 0; s < 3; ++s) {
    k_lstm<<<(NB * 32 + 255) / 256, 256, 0, stream>>>(qs, hh, cc, lwih, lwhh, lbih, lbhh);
    hipMemsetAsync(attblk, 0, (size_t)(NB + NB + NB * PD) * 4, stream);
    k_att1<<<(NN * 32 + 255) / 256, 256, 0, stream>>>(outf, hh, batch, ebuf, emax);
    k_att2<<<(NN + 255) / 256, 256, 0, stream>>>(ebuf, emax, batch, abuf, esum);
    k_att3<<<(NN * 32 + 255) / 256, 256, 0, stream>>>(outf, abuf, esum, batch, rvec);
    k_qstar<<<(NB * 2 * PD + 255) / 256, 256, 0, stream>>>(hh, rvec, qs);
  }

  k_head<<<(NB * 32 + 255) / 256, 256, 0, stream>>>(qs, l1w, l1b, l2w, l2b, lfw, lfb,
                                                    (float*)d_out);
}
